// GNNMLP_50766513439391
// MI455X (gfx1250) — matrix operand order repaired and checked by reading, not compiled, not run on hardware
//
#include <hip/hip_runtime.h>
#include <hip/hip_bf16.h>

// ---------------------------------------------------------------------------
// Fused SAGE(pool) x2 + AvgPool + per-type actor MLPs for gfx1250 (MI455X).
// - All GEMMs on v_wmma_f32_16x16x32_bf16 (f32 accumulate).
// - Weights pre-converted (bf16, transposed) into d_ws by a tiny prep kernel,
//   then DMA'd into LDS with TENSOR_LOAD_TO_LDS (TDM), double-buffered so the
//   DMA of slab i+2 overlaps the WMMAs on slab i+1. The D# pad fields insert
//   the 64->72 bf16 row padding during the DMA.
// - Block = 128 threads (4 wave32), 16 graphs (96 node rows) per block.
// ---------------------------------------------------------------------------

#define G_TOTAL 131072
#define NAG     6
#define DIM     64
#define GPB     16              // graphs per block
#define ROWS    (GPB * NAG)     // 96 node rows
#define STR     72              // LDS row stride in bf16 elements (144B)
#define TPB     128

typedef __bf16 bf16;
typedef __attribute__((ext_vector_type(16))) __bf16 v16bf;
typedef __attribute__((ext_vector_type(8)))  float  v8f;
typedef __attribute__((ext_vector_type(4)))  unsigned int u32x4;
typedef __attribute__((ext_vector_type(8)))  int i32x8;
typedef __attribute__((ext_vector_type(4)))  int i32x4;

// LDS layout (single carved allocation => known byte offsets for the TDM D#).
#define OFF_X   0u                          // sX: 96 x 72 bf16   (13824 B)
#define OFF_Y   13824u                      // sY: 96 x 72 bf16   (13824 B)
#define OFF_W0  27648u                      // weight slab 0      ( 9216 B)
#define OFF_W1  36864u                      // weight slab 1      ( 9216 B)
#define OFF_N   46080u                      // sN: 16 x 72 bf16   ( 2304 B)
#define OFF_F   48384u                      // sF: 16 x 68 f32    ( 4352 B)
#define OFF_OUT 52736u                      // sOut: 16*6*2 f32   (  768 B)
#define SMEM_BYTES 53504u

#if defined(__has_builtin)
#if __has_builtin(__builtin_amdgcn_tensor_load_to_lds)
#define USE_TDM 1
#else
#define USE_TDM 0
#endif
#else
#define USE_TDM 0
#endif

struct Frag32 { u32x4 lo, hi; };

// A fragment (16x32 bf16): lanes 0-15 row M=lane, K in {0..7,16..23}+k0;
// lanes 16-31 row M=lane-16, K in {8..15,24..31}+k0.  Two ds_load_b128.
__device__ __forceinline__ v16bf ld_frag_a(const bf16* base, int mBase, int k0, int lane) {
    const bf16* p = base + (mBase + (lane & 15)) * STR + k0 + ((lane >> 4) & 1) * 8;
    Frag32 f;
    f.lo = *(const u32x4*)(p);
    f.hi = *(const u32x4*)(p + 16);
    return __builtin_bit_cast(v16bf, f);
}

// B fragment (32x16 bf16) from transposed weight wt[n*STR + k]:
// N = lane&15, lanes 0-15 K in {0..7,16..23}+k0; lanes 16-31 K in {8..15,24..31}+k0.
__device__ __forceinline__ v16bf ld_frag_b(const bf16* wt, int n0, int k0, int lane) {
    const bf16* p = wt + (n0 + (lane & 15)) * STR + k0 + ((lane >> 4) & 1) * 8;
    Frag32 f;
    f.lo = *(const u32x4*)(p);
    f.hi = *(const u32x4*)(p + 16);
    return __builtin_bit_cast(v16bf, f);
}

__device__ __forceinline__ v8f wmma_bf(v16bf a, v16bf b, v8f c) {
    return __builtin_amdgcn_wmma_f32_16x16x32_bf16(false, a, false, b, (short)0, c,
                                                   false, false);
}

__device__ __forceinline__ v8f gemm16(const bf16* A, int mBase, const bf16* WT,
                                      int n0, int lane) {
    v8f c = {0.f, 0.f, 0.f, 0.f, 0.f, 0.f, 0.f, 0.f};
    c = wmma_bf(ld_frag_a(A, mBase, 0, lane),  ld_frag_b(WT, n0, 0, lane),  c);
    c = wmma_bf(ld_frag_a(A, mBase, 32, lane), ld_frag_b(WT, n0, 32, lane), c);
    return c;
}

#if USE_TDM
// TDM: DMA one 64x64 bf16 slab (contiguous in ws) into LDS at ldsOff, inserting
// 4 dwords of padding after every 32 dwords (row stride 64 -> 72 bf16).
__device__ __forceinline__ void tdm_load_w(const bf16* gsrc, unsigned ldsOff) {
    unsigned long long ga = (unsigned long long)(size_t)gsrc;
    u32x4 g0;
    g0[0] = 1u;                                     // count=1, user descriptor
    g0[1] = ldsOff;                                 // lds_addr (bytes)
    g0[2] = (unsigned)ga;                           // global_addr[31:0]
    g0[3] = (unsigned)((ga >> 32) & 0x1FFFFFFull)   // global_addr[56:32]
          | (2u << 30);                             // type = 2 ("image")
    i32x8 g1;
    g1[0] = (int)((1u << 16)      // data_size = 2 bytes
                | (1u << 20)      // pad_enable
                | (4u << 22)      // pad_interval: 32 dwords
                | (3u << 25));    // pad_amount:   4 dwords
    g1[1] = (int)(64u << 16);     // tensor_dim0 = 64
    g1[2] = (int)(64u << 16);     // tensor_dim1 = 64
    g1[3] = (int)(64u << 16);     // tile_dim0 = 64
    g1[4] = 64;                   // tile_dim1 = 64 (tile_dim2 = 0)
    g1[5] = 64;                   // tensor_dim0_stride = 64
    g1[6] = 0;
    g1[7] = 0;
    i32x4 z4 = {0, 0, 0, 0};
    i32x8 z8 = {0, 0, 0, 0, 0, 0, 0, 0};
    // 6-arg form (clang-23 / therock headers): (g0, g1, g2, g3, g4, cpol)
    __builtin_amdgcn_tensor_load_to_lds(g0, g1, z4, z4, z8, 0);
}
#endif

// Fetch one prepared weight slab into LDS (TDM with fallback copy).
__device__ __forceinline__ void fetch_w(const bf16* __restrict__ gsrc, unsigned ldsOff,
                                        char* smem, int tid) {
#if USE_TDM
    (void)smem; (void)tid;
    tdm_load_w(gsrc, ldsOff);           // every wave issues its own copy
#else
    bf16* dst = (bf16*)(smem + ldsOff);
    for (int i = tid; i < 64 * 64; i += TPB) {
        int n = i >> 6, k = i & 63;
        dst[n * STR + k] = gsrc[i];
    }
#endif
}

#if USE_TDM
#define WWAIT(n) __builtin_amdgcn_s_wait_tensorcnt(n)
#else
#define WWAIT(n)
#endif

__device__ __forceinline__ unsigned pk2(float a, float b) {
    unsigned short x = __builtin_bit_cast(unsigned short, (bf16)a);
    unsigned short y = __builtin_bit_cast(unsigned short, (bf16)b);
    return (unsigned)x | ((unsigned)y << 16);
}

// ---------------------------------------------------------------------------
// Prep kernel: 12 slabs of bf16-transposed 64x64 weights into ws.
// slab s, ws[s*4096 + n*64 + k] = W_s[k*64 + n].
// Order: Wp1 Wn1 Ws1 Wp2 Wn2 Ws2 M1[0] M2[0] M3[0] M1[1] M2[1] M3[1]
// ---------------------------------------------------------------------------
__global__ __launch_bounds__(256)
void prep_weights(const float* __restrict__ Wp1, const float* __restrict__ Wn1,
                  const float* __restrict__ Ws1, const float* __restrict__ Wp2,
                  const float* __restrict__ Wn2, const float* __restrict__ Ws2,
                  const float* __restrict__ MW1, const float* __restrict__ MW2,
                  const float* __restrict__ MW3, bf16* __restrict__ ws) {
    int s = blockIdx.x;
    const float* W;
    switch (s) {
        case 0:  W = Wp1;        break;
        case 1:  W = Wn1;        break;
        case 2:  W = Ws1;        break;
        case 3:  W = Wp2;        break;
        case 4:  W = Wn2;        break;
        case 5:  W = Ws2;        break;
        case 6:  W = MW1;        break;
        case 7:  W = MW2;        break;
        case 8:  W = MW3;        break;
        case 9:  W = MW1 + 4096; break;
        case 10: W = MW2 + 4096; break;
        default: W = MW3 + 4096; break;
    }
    for (int i = threadIdx.x; i < 64 * 64; i += 256) {
        int n = i >> 6, k = i & 63;
        ws[(size_t)s * 4096 + i] = (bf16)W[k * 64 + n];
    }
}

// ---------------------------------------------------------------------------
// Main fused kernel.
// ---------------------------------------------------------------------------
__global__ __launch_bounds__(TPB)
void gnn_actor_fused(const float* __restrict__ obs, const bf16* __restrict__ ws,
                     const float* __restrict__ bp1, const float* __restrict__ b1,
                     const float* __restrict__ bp2, const float* __restrict__ b2,
                     const float* __restrict__ Mb1, const float* __restrict__ Mb2,
                     const float* __restrict__ Mb3, const float* __restrict__ MW4,
                     const float* __restrict__ Mb4, float* __restrict__ out) {
    __shared__ __align__(16) char smem[SMEM_BYTES];
    bf16*  sX   = (bf16*)(smem + OFF_X);
    bf16*  sY   = (bf16*)(smem + OFF_Y);
    bf16*  sW0  = (bf16*)(smem + OFF_W0);
    bf16*  sW1  = (bf16*)(smem + OFF_W1);
    bf16*  sN   = (bf16*)(smem + OFF_N);
    float* sF   = (float*)(smem + OFF_F);
    float* sOut = (float*)(smem + OFF_OUT);

    const int tid   = threadIdx.x;
    const int lane  = tid & 31;
    const int n0    = (tid >> 5) * 16;
    const int nn    = n0 + (lane & 15);
    const int rb    = (lane >> 4) ? 8 : 0;
    const int gBase = blockIdx.x * GPB;

    // Kick off the first two weight DMAs, stage obs (f32 -> bf16) meanwhile.
    fetch_w(ws + 0 * 4096, OFF_W0, smem, tid);      // Wp1
    fetch_w(ws + 1 * 4096, OFF_W1, smem, tid);      // Wn1
    {
        const float4* src = (const float4*)(obs + (size_t)gBase * NAG * DIM);
#pragma unroll 4
        for (int i = tid; i < ROWS * 16; i += TPB) {
            float4 v = src[i];
            int row = i >> 4, c = (i & 15) * 4;
            *(uint2*)&sX[row * STR + c] = make_uint2(pk2(v.x, v.y), pk2(v.z, v.w));
        }
    }
    __syncthreads();

    // ---- L1: P = relu(X @ Wp1 + bp1) -> sY  [slab0 in W0] ------------------
    WWAIT(1);
    {
        float bb = bp1[nn];
#pragma unroll
        for (int mt = 0; mt < 6; ++mt) {
            v8f c = gemm16(sX, mt * 16, sW0, n0, lane);
            int rbase = mt * 16 + rb;
#pragma unroll
            for (int r = 0; r < 8; ++r) {
                float v = c[r] + bb;
                sY[(rbase + r) * STR + nn] = (bf16)(v > 0.f ? v : 0.f);
            }
        }
    }
    __syncthreads();
    fetch_w(ws + 2 * 4096, OFF_W0, smem, tid);      // Ws1 -> W0

    // ---- neigh = max over agents -> sN ------------------------------------
    for (int i = tid; i < GPB * DIM; i += TPB) {
        int g = i >> 6, n = i & 63;
        float m = (float)sY[(g * NAG) * STR + n];
#pragma unroll
        for (int a = 1; a < NAG; ++a) m = fmaxf(m, (float)sY[(g * NAG + a) * STR + n]);
        sN[g * STR + n] = (bf16)m;
    }
    __syncthreads();

    // ---- NN = neigh @ Wn1 -> sF  [slab1 in W1] ----------------------------
    WWAIT(1);
    {
        v8f c = gemm16(sN, 0, sW1, n0, lane);
#pragma unroll
        for (int r = 0; r < 8; ++r) sF[(rb + r) * 68 + nn] = c[r];
    }
    __syncthreads();
    fetch_w(ws + 3 * 4096, OFF_W1, smem, tid);      // Wp2 -> W1

    // ---- H1 = tanh(X @ Ws1 + NN + b1) -> sY  [slab2 in W0] ----------------
    WWAIT(1);
    {
        float bb = b1[nn];
#pragma unroll
        for (int mt = 0; mt < 6; ++mt) {
            v8f c = gemm16(sX, mt * 16, sW0, n0, lane);
            int rbase = mt * 16 + rb;
#pragma unroll
            for (int r = 0; r < 8; ++r) {
                int row = rbase + r, g = row / 6;
                sY[row * STR + nn] = (bf16)tanhf(c[r] + sF[g * 68 + nn] + bb);
            }
        }
    }
    __syncthreads();
    fetch_w(ws + 4 * 4096, OFF_W0, smem, tid);      // Wn2 -> W0

    // ---- L2: P2 = relu(H1 @ Wp2 + bp2) -> sX  [slab3 in W1] ---------------
    WWAIT(1);
    {
        float bb = bp2[nn];
#pragma unroll
        for (int mt = 0; mt < 6; ++mt) {
            v8f c = gemm16(sY, mt * 16, sW1, n0, lane);
            int rbase = mt * 16 + rb;
#pragma unroll
            for (int r = 0; r < 8; ++r) {
                float v = c[r] + bb;
                sX[(rbase + r) * STR + nn] = (bf16)(v > 0.f ? v : 0.f);
            }
        }
    }
    __syncthreads();
    fetch_w(ws + 5 * 4096, OFF_W1, smem, tid);      // Ws2 -> W1

    // ---- neigh2 -> sN ------------------------------------------------------
    for (int i = tid; i < GPB * DIM; i += TPB) {
        int g = i >> 6, n = i & 63;
        float m = (float)sX[(g * NAG) * STR + n];
#pragma unroll
        for (int a = 1; a < NAG; ++a) m = fmaxf(m, (float)sX[(g * NAG + a) * STR + n]);
        sN[g * STR + n] = (bf16)m;
    }
    __syncthreads();

    // ---- NN2 = neigh2 @ Wn2 -> sF  [slab4 in W0] --------------------------
    WWAIT(1);
    {
        v8f c = gemm16(sN, 0, sW0, n0, lane);
#pragma unroll
        for (int r = 0; r < 8; ++r) sF[(rb + r) * 68 + nn] = c[r];
    }
    __syncthreads();
    fetch_w(ws + 6 * 4096, OFF_W0, smem, tid);      // M1[0] -> W0

    // ---- H2 = H1 @ Ws2 + NN2 + b2 -> sX  [slab5 in W1] --------------------
    WWAIT(1);
    {
        float bb = b2[nn];
#pragma unroll
        for (int mt = 0; mt < 6; ++mt) {
            v8f c = gemm16(sY, mt * 16, sW1, n0, lane);
            int rbase = mt * 16 + rb;
#pragma unroll
            for (int r = 0; r < 8; ++r) {
                int row = rbase + r, g = row / 6;
                sX[row * STR + nn] = (bf16)(c[r] + sF[g * 68 + nn] + bb);
            }
        }
    }
    __syncthreads();
    fetch_w(ws + 7 * 4096, OFF_W1, smem, tid);      // M2[0] -> W1

    // ---- res = mean over agents -> sN -------------------------------------
    for (int i = tid; i < GPB * DIM; i += TPB) {
        int g = i >> 6, n = i & 63;
        float s = 0.f;
#pragma unroll
        for (int a = 0; a < NAG; ++a) s += (float)sX[(g * NAG + a) * STR + n];
        sN[g * STR + n] = (bf16)(s * (1.f / 6.f));
    }
    __syncthreads();

    // ---- actor MLPs: 16x64 tiles, slabs 6..11 alternate W0/W1 -------------
    bf16* bufA = sY;
    bf16* bufB = sY + GPB * STR;

    // t = 0
    WWAIT(1);                                           // slab6 (M1[0]) in W0
    {
        v8f c = gemm16(sN, 0, sW0, n0, lane);
        float bb = Mb1[nn];
#pragma unroll
        for (int r = 0; r < 8; ++r) {
            float v = c[r] + bb;
            bufA[(rb + r) * STR + nn] = (bf16)(v > 0.f ? v : 0.f);
        }
    }
    __syncthreads();
    fetch_w(ws + 8 * 4096, OFF_W0, smem, tid);          // M3[0] -> W0

    WWAIT(1);                                           // slab7 (M2[0]) in W1
    {
        v8f c = gemm16(bufA, 0, sW1, n0, lane);
        float bb = Mb2[nn];
#pragma unroll
        for (int r = 0; r < 8; ++r) {
            float v = c[r] + bb;
            bufB[(rb + r) * STR + nn] = (bf16)(v > 0.f ? v : 0.f);
        }
    }
    __syncthreads();
    fetch_w(ws + 9 * 4096, OFF_W1, smem, tid);          // M1[1] -> W1

    WWAIT(1);                                           // slab8 (M3[0]) in W0
    {
        v8f c = gemm16(bufB, 0, sW0, n0, lane);
        float bb = Mb3[nn];
#pragma unroll
        for (int r = 0; r < 8; ++r) {
            float v = c[r] + bb;
            bufA[(rb + r) * STR + nn] = (bf16)(v > 0.f ? v : 0.f);
        }
    }
    __syncthreads();
    fetch_w(ws + 10 * 4096, OFF_W0, smem, tid);         // M2[1] -> W0

    if (tid < 32) {                                     // head, type 0
        int g = tid >> 1, o = tid & 1;
        float acc = Mb4[o];
#pragma unroll 8
        for (int k = 0; k < 64; ++k)
            acc += (float)bufA[g * STR + k] * MW4[k * 2 + o];
        float v = tanhf(acc);
#pragma unroll
        for (int a = 0; a < 5; ++a) sOut[(g * NAG + a) * 2 + o] = v;
    }
    __syncthreads();

    // t = 1
    WWAIT(1);                                           // slab9 (M1[1]) in W1
    {
        v8f c = gemm16(sN, 0, sW1, n0, lane);
        float bb = Mb1[64 + nn];
#pragma unroll
        for (int r = 0; r < 8; ++r) {
            float v = c[r] + bb;
            bufA[(rb + r) * STR + nn] = (bf16)(v > 0.f ? v : 0.f);
        }
    }
    __syncthreads();
    fetch_w(ws + 11 * 4096, OFF_W1, smem, tid);         // M3[1] -> W1

    WWAIT(1);                                           // slab10 (M2[1]) in W0
    {
        v8f c = gemm16(bufA, 0, sW0, n0, lane);
        float bb = Mb2[64 + nn];
#pragma unroll
        for (int r = 0; r < 8; ++r) {
            float v = c[r] + bb;
            bufB[(rb + r) * STR + nn] = (bf16)(v > 0.f ? v : 0.f);
        }
    }
    __syncthreads();

    WWAIT(0);                                           // slab11 (M3[1]) in W1
    {
        v8f c = gemm16(bufB, 0, sW1, n0, lane);
        float bb = Mb3[64 + nn];
#pragma unroll
        for (int r = 0; r < 8; ++r) {
            float v = c[r] + bb;
            bufA[(rb + r) * STR + nn] = (bf16)(v > 0.f ? v : 0.f);
        }
    }
    __syncthreads();

    if (tid < 32) {                                     // head, type 1
        int g = tid >> 1, o = tid & 1;
        float acc = Mb4[2 + o];
#pragma unroll 8
        for (int k = 0; k < 64; ++k)
            acc += (float)bufA[g * STR + k] * MW4[128 + k * 2 + o];
        sOut[(g * NAG + 5) * 2 + o] = tanhf(acc);
    }
    __syncthreads();

    for (int i = tid; i < GPB * NAG * 2; i += TPB)
        out[(size_t)gBase * NAG * 2 + i] = sOut[i];
}

extern "C" void kernel_launch(void* const* d_in, const int* in_sizes, int n_in,
                              void* d_out, int out_size, void* d_ws, size_t ws_size,
                              hipStream_t stream) {
    (void)in_sizes; (void)n_in; (void)ws_size; (void)out_size;
    const float* obs = (const float*)d_in[0];
    const float* Wp1 = (const float*)d_in[1];
    const float* bp1 = (const float*)d_in[2];
    const float* Ws1 = (const float*)d_in[3];
    const float* Wn1 = (const float*)d_in[4];
    const float* b1  = (const float*)d_in[5];
    const float* Wp2 = (const float*)d_in[6];
    const float* bp2 = (const float*)d_in[7];
    const float* Ws2 = (const float*)d_in[8];
    const float* Wn2 = (const float*)d_in[9];
    const float* b2  = (const float*)d_in[10];
    const float* MW1 = (const float*)d_in[11];
    const float* Mb1 = (const float*)d_in[12];
    const float* MW2 = (const float*)d_in[13];
    const float* Mb2 = (const float*)d_in[14];
    const float* MW3 = (const float*)d_in[15];
    const float* Mb3 = (const float*)d_in[16];
    const float* MW4 = (const float*)d_in[17];
    const float* Mb4 = (const float*)d_in[18];
    float* out = (float*)d_out;
    bf16* ws = (bf16*)d_ws;     // needs 12*4096*2 = 98304 bytes

    prep_weights<<<dim3(12), dim3(256), 0, stream>>>(Wp1, Wn1, Ws1, Wp2, Wn2, Ws2,
                                                     MW1, MW2, MW3, ws);
    gnn_actor_fused<<<dim3(G_TOTAL / GPB), dim3(TPB), 0, stream>>>(
        obs, ws, bp1, b1, bp2, b2, Mb1, Mb2, Mb3, MW4, Mb4, out);
}
